// EdgeGatedMPNNLayer_38371237822903
// MI455X (gfx1250) — matrix operand order repaired and checked by reading, not compiled, not run on hardware
//
#include <hip/hip_runtime.h>
#include <math.h>

typedef __attribute__((ext_vector_type(16))) __bf16 v16bf;
typedef __attribute__((ext_vector_type(8)))  float  v8f;

#define DIM    128
#define NNODES 50000
#define NEDGES 600000
#define EPSV   1e-5f

// packed-weight fragment counts
#define WN_ELEMS (24*4*32*16)   // [128 K][384 cols] -> 24 col-tiles x 4 K-steps
#define WU_ELEMS (8*8*32*16)    // [256 K][128 cols] ->  8 col-tiles x 8 K-steps

__device__ __forceinline__ float gelu_exact(float v) {
    return 0.5f * v * (1.0f + erff(v * 0.70710678118654752f));
}

// gfx1250 async global->LDS copy (ASYNCcnt pipe); inline asm is portable
// across ROCm-7.2 and amdgpu-toolchain per the CDNA5 bridge doc.
__device__ __forceinline__ void async_ld_b128(const void* gaddr, unsigned lds_off) {
    asm volatile("global_load_async_to_lds_b128 %0, %1, off"
                 :: "v"(lds_off), "v"(gaddr) : "memory");
}
__device__ __forceinline__ void async_wait0() {
    asm volatile("s_wait_asynccnt 0x0" ::: "memory");
}

// ---------------------------------------------------------------------------
// Prep: pack W_phi / W_g1(node parts) / W_u into bf16 WMMA B-fragment layout
// (frag idx = ((ct*KS+ks)*32+lane)*16 + 2v+e; lane&15 = column, lane>>4 = K
// half h, VGPR v holds K=8h+2v,+1 (v<4) or 8h+2v+8,+1 (v>=4)).  Also fold BN into per-channel affine:
// scale = gamma*rsqrt(var+eps), shift = beta - mean*scale, and pre-scale the
// edge-feature weight rows + pack {W6*s, W7*s, Wg2} into one float4 table.
// ---------------------------------------------------------------------------
__global__ __launch_bounds__(256) void pack_weights(
    const float* __restrict__ Wphi, const float* __restrict__ Wg1,
    const float* __restrict__ Wu,
    const float* __restrict__ bnm, const float* __restrict__ bnv,
    const float* __restrict__ bng, const float* __restrict__ bnb,
    const float* __restrict__ Wg2,
    __bf16* __restrict__ WnP, __bf16* __restrict__ WuP,
    float* __restrict__ scaleO, float* __restrict__ shiftO,
    float4* __restrict__ EP)
{
    int i = blockIdx.x * 256 + threadIdx.x;
    if (i < WN_ELEMS) {
        int f    = i & 15;
        int frag = i >> 4;
        int lane = frag & 31;
        int fr2  = frag >> 5;
        int ks   = fr2 & 3;
        int ct   = fr2 >> 2;            // 0..23
        int v = f >> 1, e = f & 1;
        int k = ks * 32 + 8 * (lane >> 4) + (v < 4 ? f : f + 8);   // 0..127
        int c = ct * 16 + (lane & 15);                    // 0..383
        int mat = c >> 7;                                 // 0:phi 1:g1-dst 2:g1-src
        int j = c & 127;
        float val;
        if (mat == 0)      val = Wphi[k * DIM + j];
        else if (mat == 1) val = Wg1[k * DIM + j];
        else               val = Wg1[(128 + k) * DIM + j];
        WnP[i] = (__bf16)val;
    }
    int i2 = i - WN_ELEMS;
    if (i2 >= 0 && i2 < WU_ELEMS) {
        int f    = i2 & 15;
        int frag = i2 >> 4;
        int lane = frag & 31;
        int fr2  = frag >> 5;
        int ks   = fr2 & 7;
        int ct   = fr2 >> 3;            // 0..7
        int v = f >> 1, e = f & 1;
        int k = ks * 32 + 8 * (lane >> 4) + (v < 4 ? f : f + 8);   // 0..255
        int c = ct * 16 + (lane & 15);                    // 0..127
        WuP[i2] = (__bf16)Wu[k * DIM + c];
    }
    if (i < DIM) {
        float sc = bng[i] * rsqrtf(bnv[i] + EPSV);
        scaleO[i] = sc;
        shiftO[i] = bnb[i] - bnm[i] * sc;
        EP[i] = make_float4(sc * Wg1[256 * DIM + i], sc * Wg1[257 * DIM + i],
                            Wg2[i], 0.0f);
    }
}

__global__ __launch_bounds__(256) void zero_f32(float* __restrict__ p, int n4)
{
    int i = blockIdx.x * 256 + threadIdx.x;
    if (i < n4) ((float4*)p)[i] = make_float4(0.f, 0.f, 0.f, 0.f);
}

// ---------------------------------------------------------------------------
// Node GEMM: [N,128] x [128,384] -> XJ | A' | B'  where A'/B' carry the BN
// affine folded in (A' = scale*(x@Wg1_dst + b_g1) + shift, B' = scale*(x@
// Wg1_src)).  Tile staged global->LDS via async-load pipe, f32 in LDS,
// bf16 conversion during fragment build.  8 waves x 3 col-tiles, K=4x32.
// ---------------------------------------------------------------------------
__global__ __launch_bounds__(256) void node_gemm(
    const float* __restrict__ x, const __bf16* __restrict__ WnP,
    const float* __restrict__ bphi, const float* __restrict__ bg1,
    const float* __restrict__ scaleP, const float* __restrict__ shiftP,
    float* __restrict__ XJ, float* __restrict__ An, float* __restrict__ Bn)
{
    __shared__ float xtf[16 * 132];     // padded: rows land on distinct banks
    int tid = threadIdx.x;
    int r0  = blockIdx.x * 16;          // 3125 * 16 == 50000 exactly

#pragma unroll
    for (int t = 0; t < 2; ++t) {       // 512 float4 chunks / 256 threads
        int q   = tid + t * 256;
        int r   = q >> 5;               // 32 float4 per row
        int c4  = q & 31;
        const void* g = x + (size_t)(r0 + r) * DIM + c4 * 4;
        unsigned lds  = (unsigned)(size_t)(const void*)&xtf[r * 132 + c4 * 4];
        async_ld_b128(g, lds);
    }
    async_wait0();
    __syncthreads();

    int wave = tid >> 5, lane = tid & 31;
    int n = lane & 15, half = lane >> 4;

#pragma unroll
    for (int cti = 0; cti < 3; ++cti) {
        int ct = wave * 3 + cti;        // 0..23
        v8f acc = {};
#pragma unroll
        for (int ks = 0; ks < 4; ++ks) {
            v16bf af;
            const float* xr = &xtf[(lane & 15) * 132 + ks * 32 + 8 * half];
#pragma unroll
            for (int v = 0; v < 4; ++v) {
                af[2 * v]     = (__bf16)xr[2 * v];
                af[2 * v + 1] = (__bf16)xr[2 * v + 1];
                af[2 * v + 8] = (__bf16)xr[16 + 2 * v];
                af[2 * v + 9] = (__bf16)xr[16 + 2 * v + 1];
            }
            const __bf16* bp = WnP + (((ct * 4 + ks) * 32 + lane) << 4);
            v16bf bfr = *(const v16bf*)bp;
            acc = __builtin_amdgcn_wmma_f32_16x16x32_bf16(
                false, af, false, bfr, (short)0, acc, false, false);
        }
        int mat = ct >> 3;
        int j   = ((ct & 7) << 4) + n;
        float* op;
        float aa, bb;                   // out = aa*acc + bb
        if (mat == 0)      { op = XJ; aa = 1.0f;      bb = bphi[j]; }
        else if (mat == 1) { op = An; aa = scaleP[j];
                             bb = scaleP[j] * bg1[j] + shiftP[j]; }
        else               { op = Bn; aa = scaleP[j]; bb = 0.0f; }
#pragma unroll
        for (int v = 0; v < 8; ++v)
            op[(r0 + v + 8 * half) * DIM + j] = aa * acc[v] + bb;
    }
}

// ---------------------------------------------------------------------------
// Edge kernel: one wave per edge, lane owns 4 channels (float4 gathers).
// BN already folded:  h = A'[dst] + B'[src] + ef0*W6s + ef1*W7s  -> GELU
// gate = sigmoid(h . Wg2 + b); scatter msg = XJ[src]*(w*gate) into AGG[dst].
// Working set (~128 MB) fits the 192 MB L2 -> gathers/atomics stay on-chip.
// ---------------------------------------------------------------------------
__global__ __launch_bounds__(256) void edge_gate_scatter(
    const int* __restrict__ ei, const float* __restrict__ ewn,
    const float* __restrict__ ef, const float4* __restrict__ EP,
    const float* __restrict__ bg2,
    const float* __restrict__ An, const float* __restrict__ Bn,
    const float* __restrict__ XJ, float* __restrict__ AGG)
{
    int e = blockIdx.x * 8 + (threadIdx.x >> 5);
    if (e >= NEDGES) return;
    int lane = threadIdx.x & 31;
    int src = ei[e];
    int dst = ei[NEDGES + e];
    int j0  = lane * 4;

    float4 a4 = *(const float4*)(An + (size_t)dst * DIM + j0);
    float4 b4 = *(const float4*)(Bn + (size_t)src * DIM + j0);
    float av[4] = {a4.x, a4.y, a4.z, a4.w};
    float bv[4] = {b4.x, b4.y, b4.z, b4.w};

    float ef0 = ef[2 * e], ef1 = ef[2 * e + 1];

    float part = 0.0f;
#pragma unroll
    for (int t = 0; t < 4; ++t) {
        float4 p = EP[j0 + t];          // {W6*scale, W7*scale, Wg2, -}
        float h = av[t] + bv[t] + ef0 * p.x + ef1 * p.y;
        h = gelu_exact(h);
        part += h * p.z;
    }
#pragma unroll
    for (int m = 16; m >= 1; m >>= 1)
        part += __shfl_xor(part, m, 32);

    float gate  = 1.0f / (1.0f + expf(-(part + bg2[0])));
    float coeff = ewn[e] * gate;

    float4 xj4 = *(const float4*)(XJ + (size_t)src * DIM + j0);
    float xv[4] = {xj4.x, xj4.y, xj4.z, xj4.w};
    float* ap = AGG + (size_t)dst * DIM + j0;
#pragma unroll
    for (int t = 0; t < 4; ++t)
        unsafeAtomicAdd(ap + t, xv[t] * coeff);
}

// ---------------------------------------------------------------------------
// Node update: h = concat(x, agg) @ W_u (bf16 WMMA, K=256) -> GELU ->
// residual -> LayerNorm.  Tile staged via async-LDS loads (f32, padded
// stride 268 -> 12-bank row step, conflict-free for 16 rows).
// ---------------------------------------------------------------------------
__global__ __launch_bounds__(256) void node_update(
    const float* __restrict__ x, const float* __restrict__ AGG,
    const __bf16* __restrict__ WuP, const float* __restrict__ bu,
    const float* __restrict__ lng, const float* __restrict__ lnb,
    float* __restrict__ out)
{
    __shared__ float catf[16 * 268];    // cols 0..127 = x, 128..255 = agg
    __shared__ float hbuf[16 * 136];
    int tid = threadIdx.x;
    int r0  = blockIdx.x * 16;

#pragma unroll
    for (int t = 0; t < 2; ++t) {       // x part: 512 float4 chunks
        int q  = tid + t * 256;
        int r  = q >> 5, c4 = q & 31;
        const void* g = x + (size_t)(r0 + r) * DIM + c4 * 4;
        unsigned lds  = (unsigned)(size_t)(const void*)&catf[r * 268 + c4 * 4];
        async_ld_b128(g, lds);
    }
#pragma unroll
    for (int t = 0; t < 2; ++t) {       // agg part
        int q  = tid + t * 256;
        int r  = q >> 5, c4 = q & 31;
        const void* g = AGG + (size_t)(r0 + r) * DIM + c4 * 4;
        unsigned lds  = (unsigned)(size_t)(const void*)&catf[r * 268 + 128 + c4 * 4];
        async_ld_b128(g, lds);
    }
    async_wait0();
    __syncthreads();

    int wave = tid >> 5, lane = tid & 31;
    int n = lane & 15, half = lane >> 4;
    int ct = wave;                      // 0..7
    v8f acc = {};
#pragma unroll
    for (int ks = 0; ks < 8; ++ks) {
        v16bf af;
        const float* xr = &catf[(lane & 15) * 268 + ks * 32 + 8 * half];
#pragma unroll
        for (int v = 0; v < 4; ++v) {
            af[2 * v]     = (__bf16)xr[2 * v];
            af[2 * v + 1] = (__bf16)xr[2 * v + 1];
            af[2 * v + 8] = (__bf16)xr[16 + 2 * v];
            af[2 * v + 9] = (__bf16)xr[16 + 2 * v + 1];
        }
        const __bf16* bp = WuP + (((ct * 8 + ks) * 32 + lane) << 4);
        v16bf bfr = *(const v16bf*)bp;
        acc = __builtin_amdgcn_wmma_f32_16x16x32_bf16(
            false, af, false, bfr, (short)0, acc, false, false);
    }
    int j = ct * 16 + n;
#pragma unroll
    for (int v = 0; v < 8; ++v)
        hbuf[(v + 8 * half) * 136 + j] = acc[v] + bu[j];
    __syncthreads();

    // LayerNorm: 16 threads per row, each owns 8 channels
    int r = tid >> 4, q = tid & 15;
    float yv[8];
    float s1 = 0.f, s2 = 0.f;
#pragma unroll
    for (int k = 0; k < 8; ++k) {
        int c = q + 16 * k;
        float h = gelu_exact(hbuf[r * 136 + c]);
        float y = catf[r * 268 + c] + h;    // x value still resident in LDS
        yv[k] = y; s1 += y; s2 += y * y;
    }
#pragma unroll
    for (int m = 8; m >= 1; m >>= 1) {
        s1 += __shfl_xor(s1, m, 16);
        s2 += __shfl_xor(s2, m, 16);
    }
    float mu  = s1 * (1.0f / DIM);
    float var = s2 * (1.0f / DIM) - mu * mu;
    float rs  = rsqrtf(var + EPSV);
#pragma unroll
    for (int k = 0; k < 8; ++k) {
        int c = q + 16 * k;
        out[(r0 + r) * DIM + c] = (yv[k] - mu) * rs * lng[c] + lnb[c];
    }
}

// ---------------------------------------------------------------------------
extern "C" void kernel_launch(void* const* d_in, const int* in_sizes, int n_in,
                              void* d_out, int out_size, void* d_ws, size_t ws_size,
                              hipStream_t stream)
{
    const float* x    = (const float*)d_in[0];
    const int*   ei   = (const int*)  d_in[1];
    const float* ewn  = (const float*)d_in[2];
    const float* ef   = (const float*)d_in[3];
    // d_in[4] = batch (unused)
    const float* Wphi = (const float*)d_in[5];
    const float* bphi = (const float*)d_in[6];
    const float* Wg1  = (const float*)d_in[7];
    const float* bg1  = (const float*)d_in[8];
    const float* bng  = (const float*)d_in[9];
    const float* bnb  = (const float*)d_in[10];
    const float* bnm  = (const float*)d_in[11];
    const float* bnv  = (const float*)d_in[12];
    const float* Wg2  = (const float*)d_in[13];
    const float* bg2  = (const float*)d_in[14];
    const float* Wu   = (const float*)d_in[15];
    const float* bu   = (const float*)d_in[16];
    const float* lng  = (const float*)d_in[17];
    const float* lnb  = (const float*)d_in[18];

    char* ws = (char*)d_ws;
    const size_t MAT = (size_t)NNODES * DIM * sizeof(float);   // 25.6 MB
    float*  XJ   = (float*)(ws + 0 * MAT);
    float*  An   = (float*)(ws + 1 * MAT);
    float*  Bn   = (float*)(ws + 2 * MAT);
    float*  AGG  = (float*)(ws + 3 * MAT);
    __bf16* WnP  = (__bf16*)(ws + 4 * MAT);
    __bf16* WuP  = (__bf16*)(ws + 4 * MAT + (size_t)WN_ELEMS * sizeof(__bf16));
    char*   aux  = ws + 4 * MAT + (size_t)(WN_ELEMS + WU_ELEMS) * sizeof(__bf16);
    float*  scP  = (float*)(aux);             // 128 f
    float*  shP  = (float*)(aux + 512);       // 128 f
    float4* EP   = (float4*)(aux + 1024);     // 128 float4

    pack_weights<<<(WN_ELEMS + WU_ELEMS + 255) / 256, 256, 0, stream>>>(
        Wphi, Wg1, Wu, bnm, bnv, bng, bnb, Wg2, WnP, WuP, scP, shP, EP);
    zero_f32<<<(NNODES * DIM / 4 + 255) / 256, 256, 0, stream>>>(
        AGG, NNODES * DIM / 4);
    node_gemm<<<NNODES / 16, 256, 0, stream>>>(
        x, WnP, bphi, bg1, scP, shP, XJ, An, Bn);
    edge_gate_scatter<<<NEDGES / 8, 256, 0, stream>>>(
        ei, ewn, ef, EP, bg2, An, Bn, XJ, AGG);
    node_update<<<NNODES / 16, 256, 0, stream>>>(
        x, AGG, WuP, bu, lng, lnb, (float*)d_out);
}
